// LanguageMsgEncoder_31817117728825
// MI455X (gfx1250) — matrix operand order repaired and checked by reading, not compiled, not run on hardware
//
#include <hip/hip_runtime.h>
#include <stdint.h>

#define DD 512
#define NTOK 128
#define BATCH 2
#define M_BIG (BATCH*NTOK*NTOK)   // 32768
#define M_SMALL (BATCH*NTOK)      // 256

typedef __attribute__((ext_vector_type(8)))  float  v8f;
typedef __attribute__((ext_vector_type(16))) __bf16 v16bf;

union Frag { v16bf v; uint4 q[2]; };

__device__ __forceinline__ uint32_t f2bf_bits(float x){
  union { float f; uint32_t u; } c; c.f = x;
  uint32_t u = c.u + 0x7FFFu + ((c.u >> 16) & 1u);
  return u >> 16;
}

__device__ __forceinline__ float fast_tanh(float x){
  x = fminf(15.f, fmaxf(-15.f, x));
  float e = __expf(2.f * x);
  return (e - 1.f) / (e + 1.f);
}

// A-fragment (16x32 bf16), source row-major lda=DD.
// lanes 0-15: row=lane, elems 0..15 = K+0..15; lanes 16-31: K+16..31.
__device__ __forceinline__ v16bf load_a(const uint16_t* p /*= row*DD + kh, +ks*32 applied*/){
  Frag f;
  f.q[0] = *(const uint4*)(const void*)p;
  f.q[1] = *(const uint4*)(const void*)(p + 8);
  return f.v;
}
// B-fragment from packed weights: 32 contiguous bytes per lane.
__device__ __forceinline__ v16bf load_b(const uint32_t* p){
  Frag f;
  f.q[0] = *(const uint4*)(const void*)p;
  f.q[1] = *(const uint4*)(const void*)(p + 4);
  return f.v;
}

#define WMMA_BF16(A_, B_, C_) \
  __builtin_amdgcn_wmma_f32_16x16x32_bf16(false, (A_), false, (B_), (short)0, (C_), false, false)

// Pack f32 weight chunk W[512,512] into B-fragment layout:
// u32 idx = ((nt*16+ks)*32+lane)*8+v ; k = ks*32 + 16*(lane/16) + 2v(+1) ; n = nt*16 + lane%16
__global__ void pack_w(const float* __restrict__ W, uint32_t* __restrict__ out){
  int t    = blockIdx.x * 256 + threadIdx.x;          // 0..131071
  int v    = t & 7;
  int lane = (t >> 3) & 31;
  int ks   = (t >> 8) & 15;
  int nt   = t >> 12;
  int n     = (nt << 4) + (lane & 15);
  int kbase = (ks << 5) + ((lane >> 4) << 4) + (v << 1);
  uint32_t lo = f2bf_bits(W[(size_t)kbase * DD + n]);
  uint32_t hi = f2bf_bits(W[(size_t)(kbase + 1) * DD + n]);
  out[t] = lo | (hi << 16);
}

__global__ void f32_to_bf16(const float* __restrict__ in, uint16_t* __restrict__ out){
  int t = blockIdx.x * 256 + threadIdx.x;
  out[t] = (uint16_t)f2bf_bits(in[t]);
}

// Shared wave-level 32x64 K=512 GEMM core. acc pre-initialized by caller.
__device__ __forceinline__ void gemm_core(v8f acc[2][4], const uint16_t* __restrict__ Ain,
                                          const uint32_t* __restrict__ Bpk,
                                          int mrow0, int nt0, int lane){
  const uint16_t* a0p = Ain + (size_t)(mrow0 + (lane & 15)) * DD + ((lane >> 4) << 4);
  const uint16_t* a1p = a0p + 16 * DD;
  const uint32_t* bp  = Bpk + ((size_t)nt0 << 12) + (lane << 3);  // + ni*4096 + ks*256
#pragma unroll 4
  for (int ks = 0; ks < 16; ++ks) {
    v16bf a0 = load_a(a0p + ks * 32);
    v16bf a1 = load_a(a1p + ks * 32);
    v16bf bf[4];
#pragma unroll
    for (int ni = 0; ni < 4; ++ni) bf[ni] = load_b(bp + ni * 4096 + ks * 256);
#pragma unroll
    for (int ni = 0; ni < 4; ++ni) {
      acc[0][ni] = WMMA_BF16(a0, bf[ni], acc[0][ni]);
      acc[1][ni] = WMMA_BF16(a1, bf[ni], acc[1][ni]);
    }
  }
}

// ---------------- heavy GEMM: [32768,512]bf16 @ packedW ----------------
// grid (256, 4), block 256 (8 waves). Block M-tile = 128 rows = one (b,i), all j.
// FINAL=false: acc init = ubias[b,i,:]+ujb[b,j,:], tanh, bf16 -> Aout
// FINAL=true : acc init = bias[:], f32 -> out32
template<bool FINAL>
__global__ __launch_bounds__(256) void heavy_gemm(
    const uint16_t* __restrict__ Ain, uint16_t* __restrict__ Aout,
    const uint32_t* __restrict__ Bpk,
    const float* __restrict__ ubias, const float* __restrict__ ujb,
    const float* __restrict__ bias, float* __restrict__ out32)
{
  const int lane = threadIdx.x & 31;
  const int w    = threadIdx.x >> 5;
  const int wm   = w & 3, wn = w >> 2;
  const int T    = blockIdx.x;                 // == b*128 + i
  const int mrow0 = T * 128 + wm * 32;
  const int nt0   = blockIdx.y * 8 + wn * 4;
  const int cbase = (nt0 << 4) + (lane & 15);

  v8f acc[2][4];
  if (FINAL) {
    const float* bptr = bias + cbase;
#pragma unroll
    for (int ni = 0; ni < 4; ++ni) {
      float cc = bptr[ni * 16];
#pragma unroll
      for (int mi = 0; mi < 2; ++mi)
#pragma unroll
        for (int r = 0; r < 8; ++r) acc[mi][ni][r] = cc;
    }
  } else {
    const float* ubptr = ubias + ((size_t)T << 9) + cbase;
    const float* ujptr = ujb + ((size_t)(((T >> 7) << 7) + wm * 32 + ((lane >> 4) << 3)) << 9) + cbase;
#pragma unroll
    for (int ni = 0; ni < 4; ++ni) {
      float cc = ubptr[ni * 16];
#pragma unroll
      for (int mi = 0; mi < 2; ++mi)
#pragma unroll
        for (int r = 0; r < 8; ++r)
          acc[mi][ni][r] = cc + ujptr[(size_t)(mi * 16 + r) * DD + ni * 16];
    }
  }

  gemm_core(acc, Ain, Bpk, mrow0, nt0, lane);

#pragma unroll
  for (int mi = 0; mi < 2; ++mi) {
    size_t rbase = (size_t)(mrow0 + mi * 16 + ((lane >> 4) << 3)) * DD + cbase;
#pragma unroll
    for (int ni = 0; ni < 4; ++ni)
#pragma unroll
      for (int r = 0; r < 8; ++r) {
        size_t idx = rbase + (size_t)r * DD + ni * 16;
        float v = acc[mi][ni][r];
        if (FINAL) out32[idx] = v;
        else       Aout[idx]  = (uint16_t)f2bf_bits(fast_tanh(v));
      }
  }
}

// ---------------- small GEMM: [256,512]bf16 @ packedW ----------------
// grid (2, 4), block 256
template<bool HASBIAS, bool ACCUM, bool TANH, bool OUTBF>
__global__ __launch_bounds__(256) void small_gemm(
    const uint16_t* __restrict__ A, const uint32_t* __restrict__ Bpk,
    const float* __restrict__ bias,
    float* __restrict__ out32, uint16_t* __restrict__ outbf)
{
  const int lane = threadIdx.x & 31;
  const int w    = threadIdx.x >> 5;
  const int wm   = w & 3, wn = w >> 2;
  const int mrow0 = blockIdx.x * 128 + wm * 32;
  const int nt0   = blockIdx.y * 8 + wn * 4;
  const int cbase = (nt0 << 4) + (lane & 15);

  v8f acc[2][4];
#pragma unroll
  for (int ni = 0; ni < 4; ++ni) {
    float cc = HASBIAS ? bias[cbase + ni * 16] : 0.f;
#pragma unroll
    for (int mi = 0; mi < 2; ++mi)
#pragma unroll
      for (int r = 0; r < 8; ++r) acc[mi][ni][r] = cc;
  }

  gemm_core(acc, A, Bpk, mrow0, nt0, lane);

#pragma unroll
  for (int mi = 0; mi < 2; ++mi) {
    size_t rbase = (size_t)(mrow0 + mi * 16 + ((lane >> 4) << 3)) * DD + cbase;
#pragma unroll
    for (int ni = 0; ni < 4; ++ni)
#pragma unroll
      for (int r = 0; r < 8; ++r) {
        size_t idx = rbase + (size_t)r * DD + ni * 16;
        float v = acc[mi][ni][r];
        if (ACCUM) v += out32[idx];
        if (TANH)  v = fast_tanh(v);
        out32[idx] = v;
        if (OUTBF) outbf[idx] = (uint16_t)f2bf_bits(v);
      }
  }
}

// rel0[b,i,j,:] = bf16(tanh(ubias[b,i,:] + ujb[b,j,:]))
__global__ void build_rel0(const float* __restrict__ ubias, const float* __restrict__ ujb,
                           uint16_t* __restrict__ relbf){
  unsigned t = blockIdx.x * 256 + threadIdx.x;   // 0 .. 16777215
  int d  = t & 511;
  int j  = (t >> 9) & 127;
  int bi = t >> 16;                              // b*128 + i
  int b  = bi >> 7;
  float v = ubias[((size_t)bi << 9) + d] + ujb[((size_t)((b << 7) + j) << 9) + d];
  relbf[t] = (uint16_t)f2bf_bits(fast_tanh(v));
}

// For each (b,j,d): max / sum_exp over i (softmax axis = 1)
__global__ void softmax_stats(const float* __restrict__ rel,
                              float* __restrict__ mbuf, float* __restrict__ sbuf){
  int t = blockIdx.x * 256 + threadIdx.x;        // (b,j,d)
  int d = t & 511;
  int j = (t >> 9) & 127;
  int b = t >> 16;
  const float* p = rel + ((size_t)b * 16384 + j) * 512 + d;   // stride over i = 65536
  float m = -1e30f;
  for (int i = 0; i < 128; ++i) m = fmaxf(m, p[(size_t)i * 65536]);
  float s = 0.f;
  for (int i = 0; i < 128; ++i) s += __expf(p[(size_t)i * 65536] - m);
  mbuf[t] = m; sbuf[t] = s;
}

// rel_out[b,i,d] = sum_j rel * exp(rel - m[b,j,d]) / s[b,j,d]
__global__ void attn_out(const float* __restrict__ rel, const float* __restrict__ mbuf,
                         const float* __restrict__ sbuf, float* __restrict__ out){
  int t = blockIdx.x * 256 + threadIdx.x;        // (b,i,d)
  int d = t & 511;
  int i = (t >> 9) & 127;
  int b = t >> 16;
  const float* p  = rel  + ((size_t)(b * 128 + i) * 128) * 512 + d;  // stride over j = 512
  const float* mr = mbuf + (size_t)b * 65536 + d;
  const float* sr = sbuf + (size_t)b * 65536 + d;
  float acc = 0.f;
  for (int j = 0; j < 128; ++j) {
    float v = p[(size_t)j * 512];
    acc += v * __expf(v - mr[(size_t)j * 512]) / sr[(size_t)j * 512];
  }
  out[t] = acc;
}

extern "C" void kernel_launch(void* const* d_in, const int* in_sizes, int n_in,
                              void* d_out, int out_size, void* d_ws, size_t ws_size,
                              hipStream_t stream) {
  (void)in_sizes; (void)n_in; (void)out_size; (void)ws_size;
  const float* x0    = (const float*)d_in[0];
  const float* W_rel = (const float*)d_in[1];
  const float* b_rel = (const float*)d_in[2];
  const float* W_go  = (const float*)d_in[3];
  const float* b_go  = (const float*)d_in[4];
  const float* W_gr  = (const float*)d_in[5];
  const float* b_gr  = (const float*)d_in[6];
  const float* W_esa = (const float*)d_in[7];
  const float* b_esa = (const float*)d_in[8];

  char* ws = (char*)d_ws;
  size_t o = 0;
  auto alloc = [&](size_t bytes) -> void* {
    void* p = ws + o; o += (bytes + 255) & ~(size_t)255; return p;
  };
  uint16_t* relA  = (uint16_t*)alloc((size_t)M_BIG * DD * 2);   // 32 MB
  float*    relF  = (float*)   alloc((size_t)M_BIG * DD * 4);   // 64 MB (also hosts relB bf16)
  uint16_t* relB  = (uint16_t*)relF;
  uint16_t* xA    = (uint16_t*)alloc((size_t)M_SMALL * DD * 2);
  uint16_t* xB    = (uint16_t*)alloc((size_t)M_SMALL * DD * 2);
  uint16_t* riA   = (uint16_t*)alloc((size_t)M_SMALL * DD * 2);
  uint16_t* rjA   = (uint16_t*)alloc((size_t)M_SMALL * DD * 2);
  float* xnew32 = (float*)alloc((size_t)M_SMALL * DD * 4);
  float* ri32   = (float*)alloc((size_t)M_SMALL * DD * 4);
  float* rj32   = (float*)alloc((size_t)M_SMALL * DD * 4);
  float* ubias  = (float*)alloc((size_t)M_SMALL * DD * 4);
  float* ujb    = (float*)alloc((size_t)M_SMALL * DD * 4);
  float* mbuf   = (float*)alloc((size_t)M_SMALL * DD * 4);
  float* sbuf   = (float*)alloc((size_t)M_SMALL * DD * 4);
  uint32_t* pk_go   = (uint32_t*)alloc(131072 * 4);
  uint32_t* pk_rel1 = (uint32_t*)alloc(131072 * 4);
  uint32_t* pk_rel2 = (uint32_t*)alloc(131072 * 4);
  uint32_t* pk_gr1  = (uint32_t*)alloc(131072 * 4);
  uint32_t* pk_gr2  = (uint32_t*)alloc(131072 * 4);
  uint32_t* pk_gr3  = (uint32_t*)alloc(131072 * 4);
  uint32_t* pk_esa  = (uint32_t*)alloc(131072 * 4);

  pack_w<<<512, 256, 0, stream>>>(W_go,              pk_go);
  pack_w<<<512, 256, 0, stream>>>(W_rel,             pk_rel1);
  pack_w<<<512, 256, 0, stream>>>(W_rel + 512 * 512, pk_rel2);
  pack_w<<<512, 256, 0, stream>>>(W_gr,              pk_gr1);
  pack_w<<<512, 256, 0, stream>>>(W_gr + 512 * 512,  pk_gr2);
  pack_w<<<512, 256, 0, stream>>>(W_gr + 1024 * 512, pk_gr3);
  pack_w<<<512, 256, 0, stream>>>(W_esa,             pk_esa);
  f32_to_bf16<<<512, 256, 0, stream>>>(x0, xA);

  dim3 sg(2, 4), hg(256, 4);

  // ---- layer 0 (rank-structured: no NxN GEMM needed) ----
  small_gemm<true,  false, true,  true ><<<sg, 256, 0, stream>>>(xA,  pk_go,   b_go,  xnew32, xB);
  small_gemm<true,  false, false, true ><<<sg, 256, 0, stream>>>(xA,  pk_rel1, b_rel, ri32,   riA);
  small_gemm<false, false, false, true ><<<sg, 256, 0, stream>>>(xA,  pk_rel2, nullptr, rj32, rjA);
  small_gemm<true,  false, false, false><<<sg, 256, 0, stream>>>(xA,  pk_gr1,  b_gr,  ubias, nullptr);
  small_gemm<false, true,  false, false><<<sg, 256, 0, stream>>>(riA, pk_gr3,  nullptr, ubias, nullptr);
  small_gemm<false, false, false, false><<<sg, 256, 0, stream>>>(xA,  pk_gr2,  nullptr, ujb,  nullptr);
  small_gemm<false, true,  false, false><<<sg, 256, 0, stream>>>(rjA, pk_gr3,  nullptr, ujb,  nullptr);
  build_rel0<<<65536, 256, 0, stream>>>(ubias, ujb, relA);
  { uint16_t* t = xA; xA = xB; xB = t; }

  // ---- layers 1 and 2 ----
  uint16_t* relIn = relA;
  uint16_t* relOut = relB;
  for (int L = 1; L < 3; ++L) {
    float* xdst = (L == 2) ? (float*)d_out : xnew32;   // final x straight to output
    small_gemm<true,  false, true,  true ><<<sg, 256, 0, stream>>>(xA, pk_go,  b_go, xdst,  xB);
    small_gemm<true,  false, false, false><<<sg, 256, 0, stream>>>(xA, pk_gr1, b_gr, ubias, nullptr);
    small_gemm<false, false, false, false><<<sg, 256, 0, stream>>>(xA, pk_gr2, nullptr, ujb, nullptr);
    heavy_gemm<false><<<hg, 256, 0, stream>>>(relIn, relOut, pk_gr3, ubias, ujb, nullptr, nullptr);
    { uint16_t* t = relIn; relIn = relOut; relOut = t; }
    { uint16_t* t = xA; xA = xB; xB = t; }
  }

  // ---- final projection: rel @ W_esa + b_esa -> f32 (relIn == relA region here) ----
  heavy_gemm<true><<<hg, 256, 0, stream>>>(relIn, nullptr, pk_esa, nullptr, nullptr, b_esa, relF);

  // ---- softmax over i, weighted sum over j ----
  softmax_stats<<<512, 256, 0, stream>>>(relF, mbuf, sbuf);
  attn_out<<<512, 256, 0, stream>>>(relF, mbuf, sbuf, (float*)d_out + M_SMALL * DD);
}
